// MemoryGNN_1176821039974
// MI455X (gfx1250) — matrix operand order repaired and checked by reading, not compiled, not run on hardware
//
#include <hip/hip_runtime.h>

// ---------------------------------------------------------------------------
// CDNA5 (gfx1250) GCN: bf16 WMMA GEMMs, double-buffered LDS pipeline with
// async global->LDS weight staging (ASYNCcnt), pre-packed bf16 weights,
// fp32 atomic scatter aggregation.
// ---------------------------------------------------------------------------

typedef __attribute__((ext_vector_type(16))) __bf16 v16bf;
typedef __attribute__((ext_vector_type(8)))  float  v8f;

union Frag {
  v16bf v;
  unsigned int u[8];
};

#if __has_builtin(__builtin_amdgcn_cvt_pk_bf16_f32)
typedef __attribute__((ext_vector_type(2))) __bf16 v2bf;
__device__ __forceinline__ unsigned int pack2(float lo, float hi) {
  v2bf r = __builtin_amdgcn_cvt_pk_bf16_f32(lo, hi);   // v_cvt_pk_bf16_f32
  return __builtin_bit_cast(unsigned int, r);
}
#else
// Truncating fp32->bf16 pair pack: ~2 VALU ops (v_and_or_b32 path).
__device__ __forceinline__ unsigned int pack2(float lo, float hi) {
  return (__float_as_uint(hi) & 0xFFFF0000u) | (__float_as_uint(lo) >> 16);
}
#endif

__device__ __forceinline__ void wait_async0() {
#if __has_builtin(__builtin_amdgcn_s_wait_asynccnt)
  __builtin_amdgcn_s_wait_asynccnt(0);
#else
  asm volatile("s_wait_asynccnt 0" ::: "memory");
#endif
}

// Async copy of 16 bytes global -> LDS, tracked by ASYNCcnt (no VGPR data).
__device__ __forceinline__ void async_copy_b128(unsigned int* lds_dst,
                                                const unsigned int* gsrc) {
  unsigned lds_off = (unsigned)(unsigned long long)(void*)lds_dst;
  unsigned long long ga = (unsigned long long)(const void*)gsrc;
  asm volatile("global_load_async_to_lds_b128 %0, %1, off"
               :: "v"(lds_off), "v"(ga) : "memory");
}

#define BM 64
#define LDA_DW 20   // sA row stride in dwords (16 data + 4 pad; 20r mod 64 distinct)

// ---------------------------------------------------------------------------
// One-shot weight pre-pack: Wp[kp*N + n] = {bf16(W[2kp][n]), bf16(W[2kp+1][n])}
// Exactly the pair-interleaved image the GEMM's sB layout needs, so per-K-step
// B staging is a raw async dword copy.
// ---------------------------------------------------------------------------
__global__ void k_pack_w(const float* __restrict__ W, unsigned int* __restrict__ Wp,
                         int K, int N) {
  int i = blockIdx.x * blockDim.x + threadIdx.x;      // over (K/2)*N
  if (i < (K / 2) * N) {
    int kp = i / N, n = i % N;
    Wp[i] = pack2(W[(size_t)(2 * kp) * N + n], W[(size_t)(2 * kp + 1) * N + n]);
  }
}

// ---------------------------------------------------------------------------
// C[M,N] = A[M,K] @ W[K,N]; A fp32 in HBM, W pre-packed bf16 pairs,
// bf16 WMMA compute, fp32 accumulate.
// Block: 256 threads = 8 waves in 2 (M) x 4 (N). Block tile BM x (NT*64).
// Wave tile 32 x NT*16: 2 A-frags x NT B-frags -> 2*NT v_wmma per K-step.
// Double-buffered LDS, one barrier per K-step; B tiles staged with
// global_load_async_to_lds_b128. K % 32 == 0, N % (NT*64) == 0.
// ---------------------------------------------------------------------------
template <int NT>
__global__ __launch_bounds__(256) void gcn_gemm_bf16(
    const float* __restrict__ A, const unsigned int* __restrict__ Wp,
    float* __restrict__ C, int M, int N, int K) {
  constexpr int BN  = NT * 64;
  constexpr int LDB = BN + 4;   // pad: wave halves land in disjoint banks
  constexpr int QN  = BN / 4;   // uint4 quads per B pair-row

  __shared__ unsigned int sA[2][BM * LDA_DW]; // [row][k/2], packed bf16 (k,k+1)
  __shared__ unsigned int sB[2][16 * LDB];    // [k/2][n],   packed bf16 (k,k+1)

  const int tid  = threadIdx.x;
  const int lane = tid & 31;
  const int wave = tid >> 5;
  const int half = lane >> 4;   // 0: lanes 0-15, 1: lanes 16-31
  const int hm   = lane & 15;
  const int wm   = wave >> 2;   // 0..1  (M direction)
  const int wn   = wave & 3;    // 0..3  (N direction)

  const int m0  = blockIdx.x * BM;
  const int nb0 = blockIdx.y * BN;

  v8f acc[2][NT] = {};
  float4 fa[2];

  // ---- A slab: issue global loads into registers ----
  auto load_a = [&](int k0) {
#pragma unroll
    for (int j = 0; j < 2; ++j) {
      int q  = tid + j * 256;        // 8 quads per A row
      int r  = q >> 3;
      int kq = (q & 7) * 4;
      int row = m0 + r;
      if (row > M - 1) row = M - 1;  // clamp: keep EXEC all-ones for WMMA
      fa[j] = *(const float4*)&A[(size_t)row * K + (k0 + kq)];
    }
  };

  // ---- A slab: convert fp32 -> packed bf16 pairs into LDS buffer ----
  auto store_a = [&](int buf) {
#pragma unroll
    for (int j = 0; j < 2; ++j) {
      int q  = tid + j * 256;
      int r  = q >> 3;
      int kq = (q & 7) * 4;
      sA[buf][r * LDA_DW + (kq >> 1)]     = pack2(fa[j].x, fa[j].y);
      sA[buf][r * LDA_DW + (kq >> 1) + 1] = pack2(fa[j].z, fa[j].w);
    }
  };

  // ---- B slab: async global -> LDS copy (no VGPR data, frees registers) ----
  auto stage_b_async = [&](int k0, int buf) {
    const int kp0 = k0 >> 1;
#pragma unroll
    for (int j = 0; j < NT; ++j) {
      int q  = tid + j * 256;        // 16 pair-rows x QN quads
      int kp = q / QN;
      int nq = (q % QN) * 4;
      async_copy_b128(&sB[buf][kp * LDB + nq],
                      &Wp[(size_t)(kp0 + kp) * N + nb0 + nq]);
    }
  };

  // ---- prologue: fill buffer 0 ----
  load_a(0);
  stage_b_async(0, 0);
  store_a(0);
  wait_async0();

  int buf = 0;
  for (int k0 = 0; k0 < K; k0 += 32) {
    __syncthreads();                       // buffer `buf` now visible to all
    const bool more = (k0 + 32 < K);
    if (more) {
      load_a(k0 + 32);
      stage_b_async(k0 + 32, buf ^ 1);     // async into idle buffer
      if (k0 + 64 < K) {
        int r  = tid >> 2;                 // prefetch the A slab after next
        int ko = (tid & 3) * 8;
        int row = m0 + r;
        if (row > M - 1) row = M - 1;
        __builtin_prefetch(&A[(size_t)row * K + (k0 + 64 + ko)], 0, 1);
      }
    }

    // ---- fragments (16-bit WMMA VGPR layouts) ----
    Frag a[2], b[NT];
#pragma unroll
    for (int t = 0; t < 2; ++t) {
      const int mrow = wm * 32 + t * 16 + hm;
#pragma unroll
      for (int p = 0; p < 8; ++p) {
        // A 16x32: lanes 0-15 K=0..15, lanes 16-31 K=16..31, packed in even pairs
        int ka = 2 * p + half * 16;
        a[t].u[p] = sA[buf][mrow * LDA_DW + (ka >> 1)];
      }
    }
#pragma unroll
    for (int t = 0; t < NT; ++t) {
      const int ncol = wn * (NT * 16) + t * 16 + hm;
#pragma unroll
      for (int p = 0; p < 8; ++p) {
        // B 32x16: lanes 0-15 K=0..15, lanes 16-31 K=16..31 -> pair-row half*8+p
        b[t].u[p] = sB[buf][(half * 8 + p) * LDB + ncol];
      }
    }
#pragma unroll
    for (int mt = 0; mt < 2; ++mt)
#pragma unroll
      for (int nt = 0; nt < NT; ++nt)
        acc[mt][nt] = __builtin_amdgcn_wmma_f32_16x16x32_bf16(
            false, a[mt].v, false, b[nt].v, (short)0, acc[mt][nt], false, false);

    if (more) {
      store_a(buf ^ 1);                    // convert A into the idle buffer
      wait_async0();                       // B copies landed before next barrier
    }
    buf ^= 1;
  }

  // ---- store D: VGPR v -> row +v+half*8, lane hm -> column ----
#pragma unroll
  for (int mt = 0; mt < 2; ++mt) {
#pragma unroll
    for (int nt = 0; nt < NT; ++nt) {
#pragma unroll
      for (int v = 0; v < 8; ++v) {
        int row = m0 + wm * 32 + mt * 16 + v + half * 8;
        int col = nb0 + wn * (NT * 16) + nt * 16 + hm;
        if (row < M) C[(size_t)row * N + col] = acc[mt][nt][v];
      }
    }
  }
}

// ---------------------------------------------------------------------------
// Degree / normalization helpers
// ---------------------------------------------------------------------------
__global__ void k_fill1(float* p, int n) {
  int i = blockIdx.x * blockDim.x + threadIdx.x;
  if (i < n) p[i] = 1.0f;   // self-loop contributes 1 to every degree
}

__global__ void k_deg_accum(const int* __restrict__ dst, float* deg, int E) {
  int e = blockIdx.x * blockDim.x + threadIdx.x;
  if (e < E) atomicAdd(&deg[dst[e]], 1.0f);
}

__global__ void k_rsqrt_inplace(float* d, int n) {
  int i = blockIdx.x * blockDim.x + threadIdx.x;
  if (i < n) {
    float v = d[i];
    d[i] = (v > 0.0f) ? rsqrtf(v) : 0.0f;
  }
}

// agg[i,f] = h[i,f] * dinv[i]^2   (self-loop term; initializes agg)
__global__ void k_self_init(const float* __restrict__ h,
                            const float* __restrict__ dinv,
                            float* __restrict__ agg, int n, int Fsh) {
  long long total = (long long)n << Fsh;
  long long i = (long long)blockIdx.x * blockDim.x + threadIdx.x;
  if (i < total) {
    int node = (int)(i >> Fsh);
    float di = dinv[node];
    agg[i] = h[i] * di * di;
  }
}

// agg[dst,f] += h[src,f] * dinv[src]*dinv[dst]   (edge-parallel scatter-add)
__global__ void k_scatter(const float* __restrict__ h,
                          const int* __restrict__ srcv,
                          const int* __restrict__ dstv,
                          const float* __restrict__ dinv,
                          float* __restrict__ agg, int E, int Fsh) {
  const int F = 1 << Fsh;
  long long total = (long long)E << Fsh;
  long long stride = (long long)gridDim.x * blockDim.x;
  for (long long i = (long long)blockIdx.x * blockDim.x + threadIdx.x; i < total;
       i += stride) {
    int e = (int)(i >> Fsh);
    int f = (int)(i & (F - 1));
    int s = srcv[e];
    int d = dstv[e];
    float w = dinv[s] * dinv[d];
    atomicAdd(&agg[(size_t)d * F + f], h[(size_t)s * F + f] * w);
  }
}

// x[i,f] = (relu ? max(x+b,0) : x+b)
__global__ void k_bias_act(float* __restrict__ x, const float* __restrict__ b,
                           int n, int Fsh, int relu) {
  const int F = 1 << Fsh;
  long long total = (long long)n << Fsh;
  long long i = (long long)blockIdx.x * blockDim.x + threadIdx.x;
  if (i < total) {
    float v = x[i] + b[i & (F - 1)];
    x[i] = relu ? fmaxf(v, 0.0f) : v;
  }
}

// ---------------------------------------------------------------------------
static inline size_t align256(size_t v) { return (v + 255) & ~(size_t)255; }

extern "C" void kernel_launch(void* const* d_in, const int* in_sizes, int n_in,
                              void* d_out, int out_size, void* d_ws, size_t ws_size,
                              hipStream_t stream) {
  const float* x  = (const float*)d_in[0];
  const int*   ei = (const int*)d_in[1];
  const float* W1 = (const float*)d_in[2];
  const float* b1 = (const float*)d_in[3];
  const float* W2 = (const float*)d_in[4];
  const float* b2 = (const float*)d_in[5];
  float* out = (float*)d_out;

  const int Kin = 4096, F1 = 256, F2 = 128;
  const int F1sh = 8, F2sh = 7;
  const int Nn = in_sizes[0] / Kin;   // 50000
  const int E  = in_sizes[1] / 2;
  const int* srcv = ei;
  const int* dstv = ei + E;

  // workspace carve-up
  char* w = (char*)d_ws;
  float*        dinv = (float*)w;        w += align256((size_t)Nn * sizeof(float));
  float*        h1   = (float*)w;        w += align256((size_t)Nn * F1 * sizeof(float));
  float*        agg1 = (float*)w;        w += align256((size_t)Nn * F1 * sizeof(float));
  float*        h2   = (float*)w;        w += align256((size_t)Nn * F2 * sizeof(float));
  unsigned int* W1p  = (unsigned int*)w; w += align256((size_t)(Kin / 2) * F1 * sizeof(unsigned int));
  unsigned int* W2p  = (unsigned int*)w;

  const int TB = 256;
  const int gM = (Nn + BM - 1) / BM;

  // ---- one-shot weight packing (fp32 -> pair-interleaved bf16) ----
  {
    int n1 = (Kin / 2) * F1, n2 = (F1 / 2) * F2;
    k_pack_w<<<(n1 + TB - 1) / TB, TB, 0, stream>>>(W1, W1p, Kin, F1);
    k_pack_w<<<(n2 + TB - 1) / TB, TB, 0, stream>>>(W2, W2p, F1, F2);
  }

  // ---- normalization: deg (with self-loop) -> dinv ----
  k_fill1<<<(Nn + TB - 1) / TB, TB, 0, stream>>>(dinv, Nn);
  k_deg_accum<<<(E + TB - 1) / TB, TB, 0, stream>>>(dstv, dinv, E);
  k_rsqrt_inplace<<<(Nn + TB - 1) / TB, TB, 0, stream>>>(dinv, Nn);

  // ---- layer 1: h1 = x @ W1  (BN = 256 -> x streamed from HBM exactly once)
  gcn_gemm_bf16<4><<<dim3(gM, F1 / 256), 256, 0, stream>>>(x, W1p, h1, Nn, F1, Kin);
  {
    long long tot  = (long long)Nn << F1sh;
    long long etot = (long long)E << F1sh;
    k_self_init<<<(unsigned)((tot + TB - 1) / TB), TB, 0, stream>>>(h1, dinv, agg1, Nn, F1sh);
    k_scatter<<<(unsigned)((etot + TB - 1) / TB), TB, 0, stream>>>(h1, srcv, dstv, dinv, agg1, E, F1sh);
    k_bias_act<<<(unsigned)((tot + TB - 1) / TB), TB, 0, stream>>>(agg1, b1, Nn, F1sh, 1);
  }

  // ---- layer 2: h2 = agg1 @ W2  (BN = 128) ----
  gcn_gemm_bf16<2><<<dim3(gM, F2 / 128), 256, 0, stream>>>(agg1, W2p, h2, Nn, F2, F1);
  {
    long long tot  = (long long)Nn << F2sh;
    long long etot = (long long)E << F2sh;
    k_self_init<<<(unsigned)((tot + TB - 1) / TB), TB, 0, stream>>>(h2, dinv, out, Nn, F2sh);
    k_scatter<<<(unsigned)((etot + TB - 1) / TB), TB, 0, stream>>>(h2, srcv, dstv, dinv, out, E, F2sh);
    k_bias_act<<<(unsigned)((tot + TB - 1) / TB), TB, 0, stream>>>(out, b2, Nn, F2sh, 0);
  }
}
